// MA_Attention8_37495064494464
// MI455X (gfx1250) — compile-verified, matrix operand order checked by reading, not run on hardware
//
#include <hip/hip_runtime.h>
#include <hip/hip_bf16.h>
#include <math.h>

// MI455X (gfx1250) fused implementation.
// ws layout: soi_feats (1MB) | q (1MB) | k (1MB) | aff_weight (16MB) |
//            Wfc1T f16 (512KB) | Wfc2T f16 (16KB)  => ~19.6MB scratch.

typedef _Float16 h16;
typedef __attribute__((ext_vector_type(16))) _Float16 v16h;
typedef __attribute__((ext_vector_type(8)))  _Float16 v8h;
typedef __attribute__((ext_vector_type(8)))  float    v8f;

#define G_    16
#define DF    512     // D_FEATS
#define NPTS  512     // N
#define INDIM 1024
#define DG    32      // DF / G
#define DO_   512

// Load a 16x32 f16 fragment (A-layout, or B-from-transposed-tile) from LDS.
// Tile stored row-major [16][stride] halfs; stride*2 bytes must be 16B multiple.
// ISA 7.12.2: lane m=lane&15, kh=lane>>4; elements cover K = {kh*8..kh*8+7, 16+kh*8..+7}.
__device__ __forceinline__ v16h ldsFrag(const h16* base, int stride) {
  const int lane = threadIdx.x & 31;
  const int m  = lane & 15;
  const int kh = (lane >> 4) & 1;
  const h16* p = base + m * stride + kh * 8;
  v8h lo = *(const v8h*)(p);
  v8h hi = *(const v8h*)(p + 16);
  v16h f;
#pragma unroll
  for (int e = 0; e < 8; ++e) { f[e] = lo[e]; f[e + 8] = hi[e]; }
  return f;
}

__device__ __forceinline__ v8f wmma(v16h a, v16h b, v8f c) {
  return __builtin_amdgcn_wmma_f32_16x16x32_f16(false, a, false, b, (short)0, c,
                                                false, false);
}

// ---------------------------------------------------------------------------
// Kernel 0: one-time weight convert+transpose to f16.
//   Wfc1T[d][k] = (f16) Wfc1[k][d]   (512x512)
//   Wfc2T[g][d] = (f16) Wfc2[d][g]   (16x512)
// Removes all f32->f16 conversion and transposition from the dominant kernel's
// inner loop and halves its L2 weight traffic.
// ---------------------------------------------------------------------------
__global__ __launch_bounds__(256) void cvt_weights(
    const float* __restrict__ Wfc1, const float* __restrict__ Wfc2,
    h16* __restrict__ Wfc1T, h16* __restrict__ Wfc2T)
{
  int id = blockIdx.x * 256 + threadIdx.x;
  int stride = gridDim.x * 256;
  for (int e = id; e < DF * DF; e += stride) {
    int d = e >> 9, k = e & 511;
    Wfc1T[e] = (h16)Wfc1[k * DF + d];      // coalesced writes
  }
  for (int e = id; e < G_ * DF; e += stride) {
    int g = e >> 9, d = e & 511;
    Wfc2T[e] = (h16)Wfc2[d * G_ + g];
  }
}

// ---------------------------------------------------------------------------
// Kernel 1: generic C[MxN] = A[MxK] @ B[KxN] + bias   (fp32 in/out, f16 WMMA)
// Block tile 64(M) x 32(N), 8 waves arranged 4x2, K staged 32 at a time.
// ---------------------------------------------------------------------------
__global__ __launch_bounds__(256) void gemm_bias_f32(
    const float* __restrict__ A, const float* __restrict__ B,
    const float* __restrict__ bias, float* __restrict__ C,
    int M, int N, int K)
{
  extern __shared__ char smem[];
  h16* sA  = (h16*)smem;                         // [64][40]
  h16* sBt = (h16*)(smem + 64 * 40 * 2);         // [32][40]  (n-major)
  const int tid = threadIdx.x, lane = tid & 31, wv = tid >> 5;
  const int m0 = blockIdx.y * 64;
  const int n0 = blockIdx.x * 32;
  const int rt = wv >> 1, ct = wv & 1;
  v8f acc = {};
  for (int k0 = 0; k0 < K; k0 += 32) {
    __syncthreads();
#pragma unroll
    for (int t = 0; t < 8; ++t) {                // A chunk 64x32 (coalesced)
      int idx = tid + t * 256;
      int r = idx >> 5, c = idx & 31;
      sA[r * 40 + c] = (h16)A[(m0 + r) * K + k0 + c];
    }
#pragma unroll
    for (int t = 0; t < 4; ++t) {                // B chunk 32x32 -> transposed
      int idx = tid + t * 256;
      int kk = idx >> 5, n = idx & 31;
      sBt[n * 40 + kk] = (h16)B[(k0 + kk) * N + n0 + n];
    }
    if (k0 + 32 < K)
      __builtin_prefetch(&A[(m0 + (tid >> 3)) * K + k0 + 32 + (tid & 7) * 4], 0, 1);
    __syncthreads();
    v16h a = ldsFrag(sA + rt * 16 * 40, 40);
    v16h b = ldsFrag(sBt + ct * 16 * 40, 40);
    acc = wmma(a, b, acc);
  }
  const int col = n0 + ct * 16 + (lane & 15);
  const float bv = bias[col];
  const int rbase = m0 + rt * 16 + ((lane >> 4) << 3);
#pragma unroll
  for (int r = 0; r < 8; ++r)
    C[(rbase + r) * N + col] = acc[r] + bv;
}

// ---------------------------------------------------------------------------
// Kernel 2 (dominant): fused pos-embedding -> tanh(emb@Wfc1+b) -> @Wfc2+b.
// One block = 64 flattened (i,j) rows (constant i). emb (64x512 f16) lives in
// LDS, built once with sin/cos; fc1 runs 4 passes of 128 d-cols. Weight tiles
// arrive via gfx1250 async global->LDS b128 copies (ASYNCcnt) from the
// pre-transposed f16 Wfc1T -- no VGPR round-trip, no conversion in the loop.
// fc2 accumulation held persistently by waves 0..3.
// ---------------------------------------------------------------------------
__global__ __launch_bounds__(256) void pos_fc_kernel(
    const int* __restrict__ soi,
    const h16* __restrict__ Wfc1T, const float* __restrict__ bfc1,
    const h16* __restrict__ Wfc2T, const float* __restrict__ bfc2,
    float* __restrict__ affw)                    // [N*N][16]
{
  extern __shared__ char smem[];
  h16* emb  = (h16*)smem;                                    // [64][520]
  h16* sBt  = (h16*)(smem + 64 * 520 * 2);                   // [128][40]
  h16* pf1b = (h16*)(smem + 64 * 520 * 2 + 128 * 40 * 2);    // [64][136]
  h16* w2t  = (h16*)(smem + 64 * 520 * 2 + 128 * 40 * 2 + 64 * 136 * 2); // [16][136]
  const int tid = threadIdx.x, lane = tid & 31, wv = tid >> 5;
  const long row0 = (long)blockIdx.x * 64;
  const int i  = (int)(row0 >> 9);
  const int j0 = (int)(row0 & 511);
  // LDS byte offset of sBt (LDS aperture: low 32 bits == LDS address)
  const unsigned sBtOff = (unsigned)(uintptr_t)sBt;

  const float si = (float)soi[i * 2], ei = (float)soi[i * 2 + 1];
  const float ci = 0.5f * (si + ei);
  const float li = ei - si;
  const float lnk = 9.2103403719761836f / 128.0f;            // ln(10000)/128

  // Build emb[lr][c]; c layout: comp*256 + {sin:idx, cos:128+idx}
  for (int t = 0; t < 64; ++t) {
    int task = tid + t * 256;               // 64 rows * 2 comps * 128 idx
    int idx  = task & 127;
    int comp = (task >> 7) & 1;
    int lr   = task >> 8;
    int jj   = j0 + lr;
    float sj = (float)soi[jj * 2], ej = (float)soi[jj * 2 + 1];
    float cj = 0.5f * (sj + ej), lj = ej - sj;
    float pos = (comp == 0) ? fmaxf(fabsf((ci - cj) / li), 0.001f)
                            : __logf(lj / li);
    float arg = 100.0f * pos * __expf(-(float)idx * lnk);
    emb[lr * 520 + comp * 256 + idx]       = (h16)__sinf(arg);
    emb[lr * 520 + comp * 256 + 128 + idx] = (h16)__cosf(arg);
  }
  __syncthreads();

  v8f affc = {};                            // waves 0..3: rows wv*16..+15 x 16 g
  for (int pass = 0; pass < 4; ++pass) {
    const int d0 = pass * 128;
    v8f c0 = {}, c1 = {}, c2 = {}, c3 = {};
    for (int k0 = 0; k0 < DF; k0 += 32) {
      __syncthreads();
      // Async-copy Wfc1T rows d0..d0+127, bytes [2*k0, 2*k0+64) -> sBt[dk][0..31]
      // 128 rows x 64B = 512 x 16B segments; 2 per thread.
#pragma unroll
      for (int t = 0; t < 2; ++t) {
        int s = tid + t * 256;
        int dk = s >> 2, part = s & 3;
        unsigned ldsoff = sBtOff + (unsigned)(dk * 80 + part * 16);
        unsigned goff   = (unsigned)((((d0 + dk) * DF + k0) << 1) + part * 16);
        asm volatile("global_load_async_to_lds_b128 %0, %1, %2"
                     :: "v"(ldsoff), "v"(goff), "s"(Wfc1T) : "memory");
      }
      if (k0 + 32 < DF)
        __builtin_prefetch(Wfc1T + (d0 + (tid & 127)) * DF + k0 + 32, 0, 1);
      asm volatile("s_wait_asynccnt 0x0" ::: "memory");
      __syncthreads();
      v16h b  = ldsFrag(sBt + (wv * 16) * 40, 40);
      v16h a0 = ldsFrag(emb + 0 * 16 * 520 + k0, 520);
      v16h a1 = ldsFrag(emb + 1 * 16 * 520 + k0, 520);
      v16h a2 = ldsFrag(emb + 2 * 16 * 520 + k0, 520);
      v16h a3 = ldsFrag(emb + 3 * 16 * 520 + k0, 520);
      c0 = wmma(a0, b, c0);
      c1 = wmma(a1, b, c1);
      c2 = wmma(a2, b, c2);
      c3 = wmma(a3, b, c3);
    }
    // bias + tanh -> pf1b (64 x 128 local), then fc2 chunk accumulation
    const float bv = bfc1[d0 + wv * 16 + (lane & 15)];
    const int lc = wv * 16 + (lane & 15);
    const int rb = (lane >> 4) << 3;
    __syncthreads();                        // prior pass fc2 reads done
#pragma unroll
    for (int r = 0; r < 8; ++r) {
      pf1b[(0 * 16 + rb + r) * 136 + lc] = (h16)tanhf(c0[r] + bv);
      pf1b[(1 * 16 + rb + r) * 136 + lc] = (h16)tanhf(c1[r] + bv);
      pf1b[(2 * 16 + rb + r) * 136 + lc] = (h16)tanhf(c2[r] + bv);
      pf1b[(3 * 16 + rb + r) * 136 + lc] = (h16)tanhf(c3[r] + bv);
    }
    {   // Wfc2T chunk: 16 rows x 128 d (256B/row) -> [16][136]; 16 segs/row
      int gg = tid >> 4, pp = tid & 15;
      *(v8h*)(w2t + gg * 136 + pp * 8) =
          *(const v8h*)(Wfc2T + gg * DF + d0 + pp * 8);
    }
    __syncthreads();
    if (wv < 4) {
#pragma unroll
      for (int kc = 0; kc < 4; ++kc) {
        v16h a = ldsFrag(pf1b + (wv * 16) * 136 + kc * 32, 136);
        v16h b = ldsFrag(w2t + kc * 32, 136);
        affc = wmma(a, b, affc);
      }
    }
  }
  if (wv < 4) {
    const int g = lane & 15;
    const float bv = bfc2[g];
    const int rb = wv * 16 + ((lane >> 4) << 3);
#pragma unroll
    for (int r = 0; r < 8; ++r)
      affw[(row0 + rb + r) * G_ + g] = affc[r] + bv;
  }
}

// ---------------------------------------------------------------------------
// Kernel 3: fused attention per (16-row i-tile, group g).
//   S = Q_g @ K_gᵀ / sqrt(32); W = affw * S; P = softmax_m(W);
//   out_t = P @ soi_feats; out = out_t @ W_conv[g]ᵀ + b_conv.
// Dynamic LDS regions (total 92160B):
//   [0,1280)      qtile [16][40]h
//   [1280,42240)  regA: kgT[512][40]h -> sfT[512][40]h -> ot[16][520]h
//   [42240,75520) regB: wbuf[16][520]f -> wcv[32][520]h
//   [75520,92160) P [16][520]h
// ---------------------------------------------------------------------------
__global__ __launch_bounds__(256) void attn_kernel(
    const float* __restrict__ qb, const float* __restrict__ kb,
    const float* __restrict__ soif, const float* __restrict__ affw,
    const float* __restrict__ Wconv, const float* __restrict__ bconv,
    float* __restrict__ out)
{
  extern __shared__ char smem[];
  h16*   qtile = (h16*)smem;
  h16*   kgT   = (h16*)(smem + 1280);
  float* wbuf  = (float*)(smem + 42240);
  h16*   wcv   = (h16*)(smem + 42240);
  h16*   P     = (h16*)(smem + 75520);
  __shared__ float red[16][16];
  __shared__ float rmax[16];
  __shared__ float rinv[16];

  const int tid = threadIdx.x, lane = tid & 31, wv = tid >> 5;
  const int i0 = blockIdx.x * 16;
  const int g  = blockIdx.y;

#pragma unroll
  for (int t = 0; t < 2; ++t) {             // Q tile 16x32
    int e = tid + t * 256;
    int r = e >> 5, d = e & 31;
    qtile[r * 40 + d] = (h16)qb[(i0 + r) * DF + g * DG + d];
  }
  for (int t = 0; t < 64; ++t) {            // K_g full 512x32 (coalesced)
    int e = tid + t * 256;
    int m = e >> 5, d = e & 31;
    kgT[m * 40 + d] = (h16)kb[m * DF + g * DG + d];
  }
  __syncthreads();

  // ---- pass 1: weighted scores into wbuf ----
  const float scale = 0.17677669529663687f;   // 1/sqrt(32)
  v16h aq = ldsFrag(qtile, 40);
  for (int mt = wv; mt < 32; mt += 8) {
    v8f s = {};
    v16h bk = ldsFrag(kgT + mt * 16 * 40, 40);
    s = wmma(aq, bk, s);
    int mcol = mt * 16 + (lane & 15);
    int rb = (lane >> 4) << 3;
#pragma unroll
    for (int r = 0; r < 8; ++r) {
      int irow = rb + r;
      float aw = affw[((long)(i0 + irow) * NPTS + mcol) * G_ + g];
      wbuf[irow * 520 + mcol] = aw * (s[r] * scale);
    }
  }
  __syncthreads();

  // ---- softmax over m (16 threads per row) ----
  {
    int row = tid >> 4, part = tid & 15;
    float mx = -3.0e38f;
    for (int c = part * 32; c < part * 32 + 32; ++c)
      mx = fmaxf(mx, wbuf[row * 520 + c]);
    red[row][part] = mx;
    __syncthreads();
    if (part == 0) {
      float m2 = red[row][0];
      for (int p = 1; p < 16; ++p) m2 = fmaxf(m2, red[row][p]);
      rmax[row] = m2;
    }
    __syncthreads();
    float rm = rmax[row], sum = 0.f;
    for (int c = part * 32; c < part * 32 + 32; ++c) {
      float e = __expf(wbuf[row * 520 + c] - rm);
      wbuf[row * 520 + c] = e;
      sum += e;
    }
    red[row][part] = sum;
    __syncthreads();
    if (part == 0) {
      float s2 = 0.f;
      for (int p = 0; p < 16; ++p) s2 += red[row][p];
      rinv[row] = 1.0f / s2;
    }
    __syncthreads();
    float ri = rinv[row];
    for (int c = part * 32; c < part * 32 + 32; ++c)
      P[row * 520 + c] = (h16)(wbuf[row * 520 + c] * ri);
  }
  __syncthreads();

  // ---- pass 2: out_t = P @ soi_feats; wave owns 4 d-tiles ----
  h16* sfT = kgT;                           // reuse regA
  v8f o0 = {}, o1 = {}, o2 = {}, o3 = {};
  const int dtb = wv * 4;
  for (int m0 = 0; m0 < NPTS; m0 += 32) {
    __syncthreads();
    for (int t = 0; t < 64; ++t) {          // V chunk 32m x 512d, transposed
      int e = tid + t * 256;
      int mm = e >> 9, d = e & 511;         // coalesced global reads
      sfT[d * 40 + mm] = (h16)soif[(m0 + mm) * DF + d];
    }
    __syncthreads();
    v16h a = ldsFrag(P + m0, 520);
    o0 = wmma(a, ldsFrag(sfT + (dtb + 0) * 16 * 40, 40), o0);
    o1 = wmma(a, ldsFrag(sfT + (dtb + 1) * 16 * 40, 40), o1);
    o2 = wmma(a, ldsFrag(sfT + (dtb + 2) * 16 * 40, 40), o2);
    o3 = wmma(a, ldsFrag(sfT + (dtb + 3) * 16 * 40, 40), o3);
  }
  __syncthreads();

  // ---- pass 3: out = out_t @ Wconv[g]ᵀ + b_conv ----
  h16* ot = (h16*)(smem + 1280);            // reuse regA as [16][520]
  {
    int rb = (lane >> 4) << 3, nn = lane & 15;
#pragma unroll
    for (int r = 0; r < 8; ++r) {
      ot[(rb + r) * 520 + (dtb + 0) * 16 + nn] = (h16)o0[r];
      ot[(rb + r) * 520 + (dtb + 1) * 16 + nn] = (h16)o1[r];
      ot[(rb + r) * 520 + (dtb + 2) * 16 + nn] = (h16)o2[r];
      ot[(rb + r) * 520 + (dtb + 3) * 16 + nn] = (h16)o3[r];
    }
  }
  for (int t = 0; t < 64; ++t) {            // W_conv[g] 32o x 512d (o-major == Bt!)
    int e = tid + t * 256;
    int o = e >> 9, d = e & 511;
    wcv[o * 520 + d] = (h16)Wconv[((long)g * DG + o) * DF + d];
  }
  __syncthreads();
  if (wv < 2) {
    v8f c = {};
    for (int k0 = 0; k0 < DF; k0 += 32) {
      v16h a = ldsFrag(ot + k0, 520);
      v16h b = ldsFrag(wcv + (wv * 16) * 520 + k0, 520);
      c = wmma(a, b, c);
    }
    int ocol = wv * 16 + (lane & 15);
    float bv = bconv[g * DG + ocol];
    int rb = (lane >> 4) << 3;
#pragma unroll
    for (int r = 0; r < 8; ++r)
      out[(long)(i0 + rb + r) * DO_ + g * DG + ocol] = c[r] + bv;
  }
}

// ---------------------------------------------------------------------------
extern "C" void kernel_launch(void* const* d_in, const int* in_sizes, int n_in,
                              void* d_out, int out_size, void* d_ws, size_t ws_size,
                              hipStream_t stream) {
  const float* feats  = (const float*)d_in[0];
  const int*   soi    = (const int*)d_in[1];
  const float* W_emb  = (const float*)d_in[2];
  const float* b_emb  = (const float*)d_in[3];
  const float* W_fc1  = (const float*)d_in[4];
  const float* b_fc1  = (const float*)d_in[5];
  const float* W_fc2  = (const float*)d_in[6];
  const float* b_fc2  = (const float*)d_in[7];
  const float* W_q    = (const float*)d_in[8];
  const float* b_q    = (const float*)d_in[9];
  const float* W_k    = (const float*)d_in[10];
  const float* b_k    = (const float*)d_in[11];
  const float* W_conv = (const float*)d_in[12];
  const float* b_conv = (const float*)d_in[13];
  float* out = (float*)d_out;

  float* soif  = (float*)d_ws;                    // 512*512
  float* qbuf  = soif + NPTS * DF;
  float* kbuf  = qbuf + NPTS * DF;
  float* affw  = kbuf + NPTS * DF;                // 512*512*16
  h16*   wfc1t = (h16*)(affw + (size_t)NPTS * NPTS * G_);  // 512*512 h
  h16*   wfc2t = wfc1t + DF * DF;                 // 16*512 h

  dim3 blk(256);
  cvt_weights<<<dim3(1024), blk, 0, stream>>>(W_fc1, W_fc2, wfc1t, wfc2t);

  size_t lds1 = (size_t)(64 * 40 + 32 * 40) * 2;
  gemm_bias_f32<<<dim3(DF / 32, NPTS / 64), blk, lds1, stream>>>(
      feats, W_emb, b_emb, soif, NPTS, DF, INDIM);
  gemm_bias_f32<<<dim3(DF / 32, NPTS / 64), blk, lds1, stream>>>(
      soif, W_q, b_q, qbuf, NPTS, DF, DF);
  gemm_bias_f32<<<dim3(DF / 32, NPTS / 64), blk, lds1, stream>>>(
      soif, W_k, b_k, kbuf, NPTS, DF, DF);

  size_t lds2 = (size_t)64 * 520 * 2 + 128 * 40 * 2 + 64 * 136 * 2 + 16 * 136 * 2;
  pos_fc_kernel<<<dim3((NPTS * NPTS) / 64), blk, lds2, stream>>>(
      soi, wfc1t, b_fc1, wfc2t, b_fc2, affw);

  size_t lds3 = 92160;
  attn_kernel<<<dim3(NPTS / 16, G_), blk, lds3, stream>>>(
      qbuf, kbuf, soif, affw, W_conv, b_conv, out);
}
